// MultiheadAttention_26946624815636
// MI455X (gfx1250) — matrix operand order repaired and checked by reading, not compiled, not run on hardware
//
#include <hip/hip_runtime.h>

// Problem constants (from reference)
#define BB 4
#define SS 2048
#define DD 512
#define HH 8
#define DKK 64
#define MM (BB * SS)        // 8192 rows of activations

typedef __attribute__((ext_vector_type(16))) __bf16 v16bf;
typedef __attribute__((ext_vector_type(8)))  __bf16 v8bf;
typedef __attribute__((ext_vector_type(8)))  float  v8f;

__device__ __forceinline__ v8f wmma_bf16(v16bf a, v16bf b, v8f c) {
    return __builtin_amdgcn_wmma_f32_16x16x32_bf16(false, a, false, b, (short)0, c,
                                                   false, false);
}

// A-fragment (16x32 bf16), contiguous k order: lane (m = lane&15, h = lane>>4) holds
// K = {k0+16h .. k0+16h+15}  -> two 16B loads.
__device__ __forceinline__ v16bf load_a_frag(const __bf16* __restrict__ row, int k0, int h) {
    v8bf lo = *(const v8bf*)(row + k0 + 16 * h);
    v8bf hi = *(const v8bf*)(row + k0 + 16 * h + 8);
    return __builtin_shufflevector(lo, hi, 0, 1, 2, 3, 4, 5, 6, 7,
                                   8, 9, 10, 11, 12, 13, 14, 15);
}

// ---------------------------------------------------------------------------
// Pack W (fp32, DxD row-major) into B-fragment order so B[k][n] = W[n][k].
// ---------------------------------------------------------------------------
__global__ void __launch_bounds__(32) pack_w_kernel(const float* __restrict__ W,
                                                    __bf16* __restrict__ Wp) {
    const int lane = threadIdx.x;            // 0..31
    const int bid  = blockIdx.x;             // nt*16 + ks  (KS = 512/32 = 16)
    const int nt = bid >> 4, ks = bid & 15;
    const int n  = nt * 16 + (lane & 15);
    const int k0 = ks * 32 + 16 * (lane >> 4);
    const float* src = W + (size_t)n * DD + k0;
    __bf16* dst = Wp + ((size_t)(bid * 32 + lane)) * 16;
    v8bf a, b;
#pragma unroll
    for (int e = 0; e < 8; ++e) a[e] = (__bf16)src[e];
#pragma unroll
    for (int e = 0; e < 8; ++e) b[e] = (__bf16)src[8 + e];
    *(v8bf*)dst = a;
    *(v8bf*)(dst + 8) = b;
}

__global__ void __launch_bounds__(256) f32_to_bf16_kernel(const float* __restrict__ src,
                                                          __bf16* __restrict__ dst, int n) {
    int i = blockIdx.x * blockDim.x + threadIdx.x;
    if (i < n) dst[i] = (__bf16)src[i];
}

// ---------------------------------------------------------------------------
// GEMM: Y[m][n] = scale * (sum_k X[m][k] * W[n][k] + bias[n])
// One wave computes a 16x64 output strip (4 accumulators); ks-loop unrolled 4x.
//   MODE 0: bf16 out, head-split [B,H,S,DK]
//   MODE 1: bf16 out, head-split transposed [B,H,DK,S]
//   MODE 2: f32 out, row-major [M,D]
// ---------------------------------------------------------------------------
template <int MODE>
__global__ void __launch_bounds__(32)
gemm_wmma_kernel(const __bf16* __restrict__ X, const __bf16* __restrict__ Wp,
                 const float* __restrict__ bias, void* __restrict__ out, float scale) {
    const int KS = DD / 32;                  // 16 k-steps
    const int lane = threadIdx.x;
    const int nt0 = blockIdx.x * 4;          // first of 4 n-tiles
    const int mt  = blockIdx.y;              // 0..511
    const int h = lane >> 4, l = lane & 15;

    const __bf16* rowA = X + (size_t)(mt * 16 + l) * DD;
    v8f acc[4];
#pragma unroll
    for (int t = 0; t < 4; ++t) acc[t] = (v8f){};

#pragma unroll 4
    for (int ks = 0; ks < KS; ++ks) {
        v16bf a = load_a_frag(rowA, ks * 32, h);
        v16bf b0 = *(const v16bf*)(Wp + ((size_t)((nt0 + 0) * KS + ks) * 32 + lane) * 16);
        v16bf b1 = *(const v16bf*)(Wp + ((size_t)((nt0 + 1) * KS + ks) * 32 + lane) * 16);
        v16bf b2 = *(const v16bf*)(Wp + ((size_t)((nt0 + 2) * KS + ks) * 32 + lane) * 16);
        v16bf b3 = *(const v16bf*)(Wp + ((size_t)((nt0 + 3) * KS + ks) * 32 + lane) * 16);
        acc[0] = wmma_bf16(a, b0, acc[0]);
        acc[1] = wmma_bf16(a, b1, acc[1]);
        acc[2] = wmma_bf16(a, b2, acc[2]);
        acc[3] = wmma_bf16(a, b3, acc[3]);
    }

#pragma unroll
    for (int t = 0; t < 4; ++t) {
        const int n = (nt0 + t) * 16 + l;
        const float bn = bias[n];
#pragma unroll
        for (int r = 0; r < 8; ++r) {
            const int m = mt * 16 + r + 8 * h;
            const float v = (acc[t][r] + bn) * scale;
            if (MODE == 2) {
                ((float*)out)[(size_t)m * DD + n] = v;
            } else {
                const int b_ = m >> 11, s = m & (SS - 1);
                const int hh = n >> 6, dk = n & (DKK - 1);
                size_t idx;
                if (MODE == 0)
                    idx = ((size_t)((b_ * HH + hh) * SS + s)) * DKK + dk;
                else
                    idx = ((size_t)((b_ * HH + hh) * DKK + dk)) * SS + s;
                ((__bf16*)out)[idx] = (__bf16)v;
            }
        }
    }
}

// ---------------------------------------------------------------------------
// One 32-column j-block of causal flash attention.  MASKED=false for blocks
// fully below the diagonal (no compares/selects); MASKED=true for the
// diagonal block only.
// ---------------------------------------------------------------------------
template <bool MASKED>
__device__ __forceinline__ void attn_block(
    int jb, int qs, int l, int h, int bh, int lane,
    const __bf16* __restrict__ Kp, const __bf16* __restrict__ Vt,
    const v16bf& aQ0, const v16bf& aQ1,
    float (&mrun)[8], float (&lsum)[8], v8f (&o)[4],
    __bf16* __restrict__ Plds) {
    const float L2E = 1.44269504f;
    const float NEG = -3.0e38f;

    const __bf16* Kbase = Kp + ((size_t)bh * SS + jb) * DKK;
    const __bf16* Vbase = Vt + ((size_t)bh * DKK + l) * SS + jb + 16 * h;

    // Prefetch next j-block of K and V (in-bounds: jb+32 <= qs <= S-16).
    if (jb + 32 < qs + 16) {
        __builtin_prefetch(Kbase + (size_t)32 * DKK + (size_t)lane * DKK, 0, 1);
        __builtin_prefetch(Vt + ((size_t)bh * DKK + lane * 2) * SS + jb + 32, 0, 1);
    }

    // Scores: S = Q (16x64) x K^T (64x32), two 16x16 column tiles.
    v16bf b00 = *(const v16bf*)(Kbase + (size_t)l * DKK + 16 * h);
    v16bf b01 = *(const v16bf*)(Kbase + (size_t)l * DKK + 32 + 16 * h);
    v16bf b10 = *(const v16bf*)(Kbase + (size_t)(16 + l) * DKK + 16 * h);
    v16bf b11 = *(const v16bf*)(Kbase + (size_t)(16 + l) * DKK + 32 + 16 * h);
    v8f s0 = {}, s1 = {};
    s0 = wmma_bf16(aQ0, b00, s0);
    s1 = wmma_bf16(aQ0, b10, s1);
    s0 = wmma_bf16(aQ1, b01, s0);
    s1 = wmma_bf16(aQ1, b11, s1);

    // Issue V B-fragment loads early; consumed after the softmax below.
    v16bf bV0 = *(const v16bf*)(Vbase + (size_t)(0 * 16) * SS);
    v16bf bV1 = *(const v16bf*)(Vbase + (size_t)(1 * 16) * SS);
    v16bf bV2 = *(const v16bf*)(Vbase + (size_t)(2 * 16) * SS);
    v16bf bV3 = *(const v16bf*)(Vbase + (size_t)(3 * 16) * SS);

    // ---- Online softmax, 8 rows in parallel (row m = r + 8h) ----
    float v0[8], v1[8], tmax[8];
#pragma unroll
    for (int r = 0; r < 8; ++r) {
        if (MASKED) {
            const int row = qs + r + 8 * h;
            v0[r] = (jb + l <= row)      ? s0[r] : NEG;
            v1[r] = (jb + 16 + l <= row) ? s1[r] : NEG;
        } else {
            v0[r] = s0[r];
            v1[r] = s1[r];
        }
        tmax[r] = fmaxf(v0[r], v1[r]);
    }
    // Cross-lane max within each 16-lane half; 8 independent chains per step.
#pragma unroll
    for (int step = 1; step <= 8; step <<= 1) {
#pragma unroll
        for (int r = 0; r < 8; ++r)
            tmax[r] = fmaxf(tmax[r], __shfl_xor(tmax[r], step));
    }
    float corr[8], p0[8], p1[8], ts[8];
#pragma unroll
    for (int r = 0; r < 8; ++r) {
        const float newmax = fmaxf(mrun[r], tmax[r]);
        corr[r] = __builtin_amdgcn_exp2f((mrun[r] - newmax) * L2E);
        p0[r]   = __builtin_amdgcn_exp2f((v0[r] - newmax) * L2E);
        p1[r]   = __builtin_amdgcn_exp2f((v1[r] - newmax) * L2E);
        ts[r]   = p0[r] + p1[r];
        mrun[r] = newmax;
    }
#pragma unroll
    for (int step = 1; step <= 8; step <<= 1) {
#pragma unroll
        for (int r = 0; r < 8; ++r)
            ts[r] += __shfl_xor(ts[r], step);
    }
#pragma unroll
    for (int r = 0; r < 8; ++r) {
        lsum[r] = lsum[r] * corr[r] + ts[r];
        // Stage P (16x32, bf16) into LDS row-major for A-fragment reload.
        Plds[(r + 8 * h) * 32 + l]      = (__bf16)p0[r];
        Plds[(r + 8 * h) * 32 + 16 + l] = (__bf16)p1[r];
    }
#pragma unroll
    for (int dt = 0; dt < 4; ++dt)
#pragma unroll
        for (int r = 0; r < 8; ++r) o[dt][r] *= corr[r];

    asm volatile("s_wait_dscnt 0" ::: "memory");  // cross-lane LDS RAW

    v8bf plo = *(const v8bf*)(&Plds[l * 32 + 16 * h]);
    v8bf phi = *(const v8bf*)(&Plds[l * 32 + 16 * h + 8]);
    v16bf aP = __builtin_shufflevector(plo, phi, 0, 1, 2, 3, 4, 5, 6, 7,
                                       8, 9, 10, 11, 12, 13, 14, 15);

    // O += P (16x32) x V (32x64); Vt is [DK][S] so B loads are contiguous.
    o[0] = wmma_bf16(aP, bV0, o[0]);
    o[1] = wmma_bf16(aP, bV1, o[1]);
    o[2] = wmma_bf16(aP, bV2, o[2]);
    o[3] = wmma_bf16(aP, bV3, o[3]);

    asm volatile("s_wait_dscnt 0" ::: "memory");  // keep next-iter stores ordered
}

// ---------------------------------------------------------------------------
// Causal flash attention: one wave handles 16 query rows of one (b,h).
// Q (pre-scaled by 1/sqrt(DK)), K: [B,H,S,DK] bf16.  Vt: [B,H,DK,S] bf16.
// Out: [B,S,D] bf16.
// ---------------------------------------------------------------------------
__global__ void __launch_bounds__(32)
attn_kernel(const __bf16* __restrict__ Qp, const __bf16* __restrict__ Kp,
            const __bf16* __restrict__ Vt, __bf16* __restrict__ Xo) {
    __shared__ __bf16 Plds[16 * 32];

    const int lane = threadIdx.x;
    const int h = lane >> 4, l = lane & 15;
    const int bh = blockIdx.x;               // 0..31
    const int qs = blockIdx.y * 16;          // query tile start
    const float NEG = -3.0e38f;

    const __bf16* Qrow = Qp + ((size_t)bh * SS + qs + l) * DKK;
    const v16bf aQ0 = load_a_frag(Qrow, 0, h);
    const v16bf aQ1 = load_a_frag(Qrow, 32, h);

    float mrun[8], lsum[8];
    v8f o[4];
#pragma unroll
    for (int r = 0; r < 8; ++r) { mrun[r] = NEG; lsum[r] = 0.0f; }
#pragma unroll
    for (int dt = 0; dt < 4; ++dt) o[dt] = (v8f){};

    int jb = 0;
    // Bulk: blocks fully below the diagonal -> no masking ops at all.
    for (; jb + 31 <= qs; jb += 32)
        attn_block<false>(jb, qs, l, h, bh, lane, Kp, Vt, aQ0, aQ1, mrun, lsum, o, Plds);
    // Diagonal block(s): per-element causal mask.
    for (; jb < qs + 16; jb += 32)
        attn_block<true>(jb, qs, l, h, bh, lane, Kp, Vt, aQ0, aQ1, mrun, lsum, o, Plds);

    // Normalize (rcp + mul instead of 32 divides) and write [B,S,D].
    float inv[8];
#pragma unroll
    for (int r = 0; r < 8; ++r) inv[r] = __builtin_amdgcn_rcpf(lsum[r]);

    const int b_ = bh >> 3, hh = bh & 7;
#pragma unroll
    for (int dt = 0; dt < 4; ++dt) {
#pragma unroll
        for (int r = 0; r < 8; ++r) {
            const float val = o[dt][r] * inv[r];
            const int srow = qs + r + 8 * h;
            const int d = hh * DKK + dt * 16 + l;
            Xo[((size_t)b_ * SS + srow) * DD + d] = (__bf16)val;
        }
    }
}

// ---------------------------------------------------------------------------
extern "C" void kernel_launch(void* const* d_in, const int* in_sizes, int n_in,
                              void* d_out, int out_size, void* d_ws, size_t ws_size,
                              hipStream_t stream) {
    const float* q  = (const float*)d_in[0];
    const float* k  = (const float*)d_in[1];
    const float* v  = (const float*)d_in[2];
    // d_in[3] = causal mask (int32) -- mask is analytic causal, applied in-kernel.
    const float* Wq = (const float*)d_in[4];  const float* bq = (const float*)d_in[5];
    const float* Wk = (const float*)d_in[6];  const float* bk = (const float*)d_in[7];
    const float* Wv = (const float*)d_in[8];  const float* bv = (const float*)d_in[9];
    const float* Wo = (const float*)d_in[10]; const float* bo = (const float*)d_in[11];

    char* ws = (char*)d_ws;
    size_t off = 0;
    auto alloc = [&](size_t bytes) -> char* {
        char* p = ws + off;
        off += (bytes + 255) & ~(size_t)255;
        return p;
    };
    const size_t actB = (size_t)MM * DD * sizeof(__bf16);   // 8 MB
    const size_t wB   = (size_t)DD * DD * sizeof(__bf16);   // 0.5 MB
    __bf16* Xq  = (__bf16*)alloc(actB);
    __bf16* Xk  = (__bf16*)alloc(actB);
    __bf16* Xv  = (__bf16*)alloc(actB);
    __bf16* Wqp = (__bf16*)alloc(wB);
    __bf16* Wkp = (__bf16*)alloc(wB);
    __bf16* Wvp = (__bf16*)alloc(wB);
    __bf16* Wop = (__bf16*)alloc(wB);
    __bf16* Qp  = (__bf16*)alloc(actB);
    __bf16* Kp  = (__bf16*)alloc(actB);
    __bf16* Vt  = (__bf16*)alloc(actB);
    __bf16* Xa  = (__bf16*)alloc(actB);

    const int nelem = MM * DD;
    f32_to_bf16_kernel<<<(nelem + 255) / 256, 256, 0, stream>>>(q, Xq, nelem);
    f32_to_bf16_kernel<<<(nelem + 255) / 256, 256, 0, stream>>>(k, Xk, nelem);
    f32_to_bf16_kernel<<<(nelem + 255) / 256, 256, 0, stream>>>(v, Xv, nelem);

    pack_w_kernel<<<512, 32, 0, stream>>>(Wq, Wqp);
    pack_w_kernel<<<512, 32, 0, stream>>>(Wk, Wkp);
    pack_w_kernel<<<512, 32, 0, stream>>>(Wv, Wvp);
    pack_w_kernel<<<512, 32, 0, stream>>>(Wo, Wop);

    const float qscale = 0.125f;               // 1/sqrt(DK) folded into Q
    dim3 gg(DD / 64, MM / 16);                 // 8 x 512 strips of 16x64
    gemm_wmma_kernel<0><<<gg, 32, 0, stream>>>(Xq, Wqp, bq, Qp, qscale);
    gemm_wmma_kernel<0><<<gg, 32, 0, stream>>>(Xk, Wkp, bk, Kp, 1.0f);
    gemm_wmma_kernel<1><<<gg, 32, 0, stream>>>(Xv, Wvp, bv, Vt, 1.0f);

    dim3 ga(BB * HH, SS / 16);                 // 32 x 128
    attn_kernel<<<ga, 32, 0, stream>>>(Qp, Kp, Vt, Xa);

    gemm_wmma_kernel<2><<<gg, 32, 0, stream>>>(Xa, Wop, bo, d_out, 1.0f);
}
